// SS2D_88519275970979
// MI455X (gfx1250) — matrix operand order repaired and checked by reading, not compiled, not run on hardware
//
#include <hip/hip_runtime.h>
#include <hip/hip_bf16.h>

// ---------------------------------------------------------------------------
// SS2D (VMamba selective scan 2D) for MI455X / gfx1250, wave32 + WMMA bf16.
// B=8, H=W=64, C=96, D=192, N_STATE=16, R=6, K=4, L=4096.
// ---------------------------------------------------------------------------

#define BB 8
#define HH 64
#define WW 64
#define CC 96
#define DD 192
#define NS 16
#define RR 6
#define KK 4
#define LL 4096

typedef __attribute__((ext_vector_type(16))) __bf16 v16bf;
typedef __attribute__((ext_vector_type(4)))  __bf16 v4bf;
typedef __attribute__((ext_vector_type(8)))  float  v8f;
typedef __attribute__((ext_vector_type(4)))  float  v4f;

// ---------------------------------------------------------------------------
// Generic tiled WMMA GEMM:  C[m,n] = sum_k A[m,k] * W[n,k]
//   KCONTIG=true : A row-major with contiguous k (csA==1), rsA = row stride.
//   KCONTIG=false: A with contiguous m (rsA==1), csA = k stride (transposed
//                  view used for x_proj where A = xs^T).
//   W: float, (N x K) row-major             (+ (z%wMod)*wBatchStride)
//   C: float, row-major ldc                 (+ z*cBatchStride)
// Block = 128 threads (4 waves); block computes a 64(M) x 16(N) tile; each
// wave owns one 16x16 C tile, looping K in chunks of 32 through LDS-staged
// bf16 operands and v_wmma_f32_16x16x32_bf16 (f32 accumulate).
// Staging uses float4 global loads (global_load_b128) + packed bf16 converts.
// Requires M % 64 == 0, K % 32 == 0 (true for all uses here); N guarded.
// ---------------------------------------------------------------------------
template<bool KCONTIG>
__global__ __launch_bounds__(128)
void gemm_bf16_wmma(const float* __restrict__ A, long long rsA, long long csA,
                    long long aBatchStride,
                    const float* __restrict__ W, int wMod, long long wBatchStride,
                    float* __restrict__ C, int ldc, long long cBatchStride,
                    int N, int K)
{
    __shared__ __align__(16) __bf16 sA[64 * 32];   // 4 KB
    __shared__ __align__(16) __bf16 sW[16 * 32];   // 1 KB

    const int z = blockIdx.z;
    A += (long long)z * aBatchStride;
    W += (long long)(z % wMod) * wBatchStride;
    C += (long long)z * cBatchStride;

    const int m0  = blockIdx.x * 64;
    const int n0  = blockIdx.y * 16;
    const int tid = threadIdx.x;
    const int wave = tid >> 5;       // 0..3 -> M sub-tile
    const int lane = tid & 31;
    const int hi = lane >> 4;        // half-wave select
    const int lo = lane & 15;

    v8f acc = {};

    for (int k0 = 0; k0 < K; k0 += 32) {
        if (KCONTIG) {
            // A tile 64x32: 512 float4 quads, 4 per thread, k-contiguous.
            #pragma unroll
            for (int q = 0; q < 4; ++q) {
                int i  = (tid + q * 128) * 4;          // element index
                int mm = i >> 5, kk = i & 31;
                v4f f = *(const v4f*)&A[(long long)(m0 + mm) * rsA + (k0 + kk)];
                *(v4bf*)&sA[mm * 32 + kk] = __builtin_convertvector(f, v4bf);
            }
        } else {
            // A tile via transposed view: float4 along contiguous m.
            #pragma unroll
            for (int q = 0; q < 4; ++q) {
                int i  = tid + q * 128;                // quad index, 512 total
                int kk = i >> 4;                       // 0..31
                int mm = (i & 15) * 4;                 // 0..60
                v4f f = *(const v4f*)&A[(long long)(m0 + mm) + (long long)(k0 + kk) * csA];
                v4bf bf = __builtin_convertvector(f, v4bf);
                sA[(mm + 0) * 32 + kk] = bf[0];
                sA[(mm + 1) * 32 + kk] = bf[1];
                sA[(mm + 2) * 32 + kk] = bf[2];
                sA[(mm + 3) * 32 + kk] = bf[3];
            }
        }
        {
            // W tile 16x32: 128 quads, 1 per thread; zero-pad rows n >= N.
            int i  = tid * 4;
            int nn = i >> 5, kk = i & 31;
            v4f f = {0.0f, 0.0f, 0.0f, 0.0f};
            if (n0 + nn < N)
                f = *(const v4f*)&W[(long long)(n0 + nn) * K + (k0 + kk)];
            *(v4bf*)&sW[nn * 32 + kk] = __builtin_convertvector(f, v4bf);
        }
        __syncthreads();

        // Gather fragments per CDNA5 16-bit WMMA VGPR layouts (ISA 7.12.2);
        // per-lane these are contiguous 16B runs -> ds_load_b128.
        v16bf a, b;
        const int arow = wave * 16 + lo;   // A: m = lane&15 within wave tile
        #pragma unroll
        for (int e = 0; e < 16; ++e) {
            // A 16x32: K = (e<8 ? 0 : 16) + 8*hi + (e&7)
            int ka = ((e & 8) << 1) + hi * 8 + (e & 7);
            a[e] = sA[arow * 32 + ka];
            // B 32x16: K = (e<8 ? 0 : 16) + 8*hi + (e&7), N = lane&15
            int kb = ((e & 8) << 1) + hi * 8 + (e & 7);
            b[e] = sW[lo * 32 + kb];
        }
        acc = __builtin_amdgcn_wmma_f32_16x16x32_bf16(
                  false, a, false, b, (short)0, acc, false, false);
        __syncthreads();
    }

    // C/D 16x16 f32 layout: VGPR j -> M = 8*hi + j, N = lane&15.
    const int col = n0 + lo;
    if (col < N) {
        #pragma unroll
        for (int j = 0; j < 8; ++j) {
            int row = m0 + wave * 16 + hi * 8 + j;
            C[(long long)row * ldc + col] = acc[j];
        }
    }
}

// ---------------------------------------------------------------------------
// Depthwise 3x3 conv (SAME, zero pad) + bias + SiLU, fused with the 4-way
// cross-scan scatter:  xs[b,k,d,l] for k in {row, col, rev-row, rev-col}.
// Reads conv input directly from xz[b, l, d] (cols 0..191 of in_proj output).
// ---------------------------------------------------------------------------
__global__ __launch_bounds__(256)
void conv_silu_scan_kernel(const float* __restrict__ xz,
                           const float* __restrict__ cw,
                           const float* __restrict__ cb,
                           float* __restrict__ xs)
{
    int idx = blockIdx.x * 256 + threadIdx.x;      // over B*D*H*W
    int w = idx & (WW - 1);
    int h = (idx >> 6) & (HH - 1);
    int d = (idx >> 12) % DD;
    int b = idx / (DD * LL);

    float acc = cb[d];
    #pragma unroll
    for (int dh = -1; dh <= 1; ++dh) {
        #pragma unroll
        for (int dw = -1; dw <= 1; ++dw) {
            int hh = h + dh, ww = w + dw;
            if (hh >= 0 && hh < HH && ww >= 0 && ww < WW)
                acc += cw[d * 9 + (dh + 1) * 3 + (dw + 1)]
                     * xz[((long long)b * LL + hh * WW + ww) * (2 * DD) + d];
        }
    }
    float v = acc / (1.0f + __expf(-acc));         // SiLU

    int l0 = h * WW + w;                           // row-major position
    int l1 = w * HH + h;                           // col-major position
    long long base = ((long long)b * KK * DD + d) * LL;
    const long long ks = (long long)DD * LL;       // per-direction stride
    xs[base + 0 * ks + l0]            = v;
    xs[base + 1 * ks + l1]            = v;
    xs[base + 2 * ks + (LL - 1 - l0)] = v;
    xs[base + 3 * ks + (LL - 1 - l1)] = v;
}

// ---------------------------------------------------------------------------
// dt projection (K=6 contraction, scalar) + softplus -> dts[b,k,d,l].
// xdblT layout: [b, k, l, 38] with c = 0..5 dt, 6..21 B, 22..37 C.
// ---------------------------------------------------------------------------
__global__ __launch_bounds__(256)
void dt_softplus_kernel(const float* __restrict__ xdblT,
                        const float* __restrict__ dtw,
                        const float* __restrict__ dtb,
                        float* __restrict__ dts)
{
    long long idx = (long long)blockIdx.x * 256 + threadIdx.x;  // (b,k,d,l)
    int l = (int)(idx & (LL - 1));
    int d = (int)((idx >> 12) % DD);
    long long bk = idx / ((long long)DD * LL);                  // b*K + k
    int k = (int)(bk & (KK - 1));

    const float* xrow = xdblT + (bk * LL + l) * 38;
    const float* wrow = dtw + ((long long)k * DD + d) * RR;
    float acc = dtb[k * DD + d];
    #pragma unroll
    for (int r = 0; r < RR; ++r) acc += wrow[r] * xrow[r];
    dts[idx] = (acc > 20.0f) ? acc : __logf(1.0f + __expf(acc));
}

// ---------------------------------------------------------------------------
// Selective scan. One thread per (b,k,d,n): 98,304 threads total, each holds
// one scalar state and iterates l = 0..4095. The n=16 reduction for
// y = sum_n h*C runs via shfl_xor within width-16 groups (two d per wave32).
// Block = 256 threads = 16 d x 16 n.
// ---------------------------------------------------------------------------
__global__ __launch_bounds__(256)
void scan_kernel(const float* __restrict__ xs, const float* __restrict__ dts,
                 const float* __restrict__ xdblT,
                 const float* __restrict__ Alogs, const float* __restrict__ Ds,
                 float* __restrict__ ys)
{
    int dblk = blockIdx.x % (DD / 16);
    long long bk = blockIdx.x / (DD / 16);       // b*K + k
    int k = (int)(bk & (KK - 1));
    int n = threadIdx.x & 15;
    int d = dblk * 16 + (threadIdx.x >> 4);
    int kd = k * DD + d;

    const float Acoef = -__expf(Alogs[kd * NS + n]);   // A = -exp(A_log)
    const float Dskip = Ds[kd];
    const long long xbase = (bk * DD + d) * LL;
    const long long cbase = bk * (long long)LL * 38;

    float h = 0.0f;
    for (int l = 0; l < LL; ++l) {
        // Prefetch upcoming B/C row into cache (global_prefetch_b8).
        __builtin_prefetch(&xdblT[cbase + (long long)(l + 8) * 38 + 6 + n], 0, 0);

        float xt = xs[xbase + l];
        float dt = dts[xbase + l];
        float Bv = xdblT[cbase + (long long)l * 38 + 6 + n];
        float Cv = xdblT[cbase + (long long)l * 38 + 22 + n];

        float dA = __expf(dt * Acoef);
        h = h * dA + (dt * xt) * Bv;
        float part = h * Cv;
        part += __shfl_xor(part, 1, 16);
        part += __shfl_xor(part, 2, 16);
        part += __shfl_xor(part, 4, 16);
        part += __shfl_xor(part, 8, 16);
        if (n == 0) ys[xbase + l] = part + Dskip * xt;
    }
}

// ---------------------------------------------------------------------------
// Cross-merge (un-reverse / un-transpose, sum 4 dirs) + LayerNorm over D +
// SiLU(z) gate, fused. Block = 192 threads = one (b,l) site.
// ---------------------------------------------------------------------------
__global__ __launch_bounds__(192)
void merge_norm_gate_kernel(const float* __restrict__ ys,
                            const float* __restrict__ xz,
                            const float* __restrict__ nw,
                            const float* __restrict__ nb,
                            float* __restrict__ yg)
{
    __shared__ float red[12];
    int bl = blockIdx.x;                 // b*L + l
    int b = bl >> 12;
    int l = bl & (LL - 1);
    int h = l >> 6, w = l & (WW - 1);
    int lt = w * HH + h;                 // col-major twin position
    int d = threadIdx.x;

    long long base = ((long long)b * KK * DD + d) * LL;
    const long long ks = (long long)DD * LL;
    float v = ys[base + 0 * ks + l]
            + ys[base + 2 * ks + (LL - 1 - l)]
            + ys[base + 1 * ks + lt]
            + ys[base + 3 * ks + (LL - 1 - lt)];

    // Block reduction for mean / E[y^2] over D=192 (6 waves).
    float s = v, s2 = v * v;
    for (int off = 16; off >= 1; off >>= 1) {
        s  += __shfl_xor(s,  off, 32);
        s2 += __shfl_xor(s2, off, 32);
    }
    int wv = threadIdx.x >> 5, ln = threadIdx.x & 31;
    if (ln == 0) { red[wv] = s; red[6 + wv] = s2; }
    __syncthreads();
    if (threadIdx.x == 0) {
        float ts = 0.0f, ts2 = 0.0f;
        for (int i = 0; i < 6; ++i) { ts += red[i]; ts2 += red[6 + i]; }
        red[0] = ts; red[6] = ts2;
    }
    __syncthreads();
    float mu  = red[0] / (float)DD;
    float var = red[6] / (float)DD - mu * mu;

    float yn = (v - mu) * rsqrtf(var + 1e-5f) * nw[d] + nb[d];
    float zv = xz[(long long)bl * (2 * DD) + DD + d];
    float g  = zv / (1.0f + __expf(-zv));            // SiLU gate
    yg[(long long)bl * DD + d] = yn * g;
}

// ---------------------------------------------------------------------------
// Host launch
// ---------------------------------------------------------------------------
extern "C" void kernel_launch(void* const* d_in, const int* in_sizes, int n_in,
                              void* d_out, int out_size, void* d_ws, size_t ws_size,
                              hipStream_t stream)
{
    const float* x       = (const float*)d_in[0];   // (B,H,W,C)
    const float* inw     = (const float*)d_in[1];   // (384,96)
    const float* convw   = (const float*)d_in[2];   // (192,1,3,3)
    const float* convb   = (const float*)d_in[3];   // (192,)
    const float* xprojw  = (const float*)d_in[4];   // (4,38,192)
    const float* dtw     = (const float*)d_in[5];   // (4,192,6)
    const float* dtb     = (const float*)d_in[6];   // (4,192)
    const float* alogs   = (const float*)d_in[7];   // (768,16)
    const float* ds      = (const float*)d_in[8];   // (768,)
    const float* normw   = (const float*)d_in[9];   // (192,)
    const float* normb   = (const float*)d_in[10];  // (192,)
    const float* outw    = (const float*)d_in[11];  // (96,192)
    float* out           = (float*)d_out;           // (B,H,W,C)

    // Workspace carving (floats).
    float* ws    = (float*)d_ws;
    float* xz    = ws;                                       // B*L*384
    float* xs    = xz    + (long long)BB * LL * 2 * DD;      // B*K*D*L
    float* xdblT = xs    + (long long)BB * KK * DD * LL;     // B*K*L*38
    float* dts   = xdblT + (long long)BB * KK * LL * 38;     // B*K*D*L
    float* ysb   = dts   + (long long)BB * KK * DD * LL;     // B*K*D*L
    float* yg    = xs;   // xs dead after scan -> reuse for gated output

    const long long M_BL = (long long)BB * LL;               // 32768

    // 1) in_proj: xz = x @ W_in^T  (32768 x 384, K=96), WMMA bf16.
    gemm_bf16_wmma<true><<<dim3(M_BL / 64, (2 * DD) / 16, 1), 128, 0, stream>>>(
        x, (long long)CC, 1LL, 0LL,
        inw, 1, 0LL,
        xz, 2 * DD, 0LL,
        2 * DD, CC);

    // 2) depthwise conv + SiLU + 4-direction cross-scan scatter.
    conv_silu_scan_kernel<<<(BB * DD * LL) / 256, 256, 0, stream>>>(
        xz, convw, convb, xs);

    // 3) x_proj (batched over b,k = 32): xdblT[l,c] = sum_d xs[d,l]*W[k][c,d].
    //    A view of xs^T via strides rsA=1 (m=l), csA=L (k=d). N=38 guarded.
    gemm_bf16_wmma<false><<<dim3(LL / 64, 3, BB * KK), 128, 0, stream>>>(
        xs, 1LL, (long long)LL, (long long)DD * LL,
        xprojw, KK, (long long)38 * DD,
        xdblT, 38, (long long)LL * 38,
        38, DD);

    // 4) dt projection + softplus.
    dt_softplus_kernel<<<(int)(((long long)BB * KK * DD * LL) / 256), 256, 0, stream>>>(
        xdblT, dtw, dtb, dts);

    // 5) selective scan (thread per (b,k,d,n), shfl reduction over n).
    scan_kernel<<<BB * KK * (DD / 16), 256, 0, stream>>>(
        xs, dts, xdblT, alogs, ds, ysb);

    // 6) cross-merge + LayerNorm + SiLU gate.
    merge_norm_gate_kernel<<<BB * LL, DD, 0, stream>>>(
        ysb, xz, normw, normb, yg);

    // 7) out_proj: out = yg @ W_out^T  (32768 x 96, K=192), WMMA bf16.
    gemm_bf16_wmma<true><<<dim3(M_BL / 64, CC / 16, 1), 128, 0, stream>>>(
        yg, (long long)DD, 1LL, 0LL,
        outw, 1, 0LL,
        out, CC, 0LL,
        CC, DD);
}
